// LMAttention_88948772700248
// MI455X (gfx1250) — matrix operand order repaired and checked by reading, not compiled, not run on hardware
//
#include <hip/hip_runtime.h>
#include <hip/hip_bf16.h>

typedef _Float16 h8   __attribute__((ext_vector_type(8)));
typedef _Float16 v16h __attribute__((ext_vector_type(16)));
typedef float    v8f  __attribute__((ext_vector_type(8)));
typedef unsigned int v4u __attribute__((ext_vector_type(4)));
typedef int      v4i  __attribute__((ext_vector_type(4)));
typedef int      v8i  __attribute__((ext_vector_type(8)));

#define T_SEQ 2048
#define DIM   3072
#define HQ    32
#define HKV   8
#define HD    128
#define QKD   (HQ * HD)   // 4096
#define KVD   (HKV * HD)  // 1024

// Tensor Data Mover availability (probe-confirmed on both toolchains; arity differs)
#if defined(__has_builtin)
#  if __has_builtin(__builtin_amdgcn_tensor_load_to_lds) && __has_builtin(__builtin_amdgcn_s_wait_tensorcnt)
#    define USE_TDM 1
#  else
#    define USE_TDM 0
#  endif
#else
#  define USE_TDM 0
#endif

// ---------------------------------------------------------------------------
// WMMA wrapper: D = A(16x32 f16) * B(32x16 f16) + C(16x16 f32)
// ---------------------------------------------------------------------------
__device__ __forceinline__ v8f wmma32(v16h a, v16h b, v8f c) {
  return __builtin_amdgcn_wmma_f32_16x16x32_f16(
      /*neg_a=*/false, a, /*neg_b=*/false, b,
      /*c_mod=*/(short)0, c, /*reuse_a=*/false, /*reuse_b=*/false);
}

// A-fragment (16x32, M x K): lane m (m = lane&15) holds row m.
// Lane-half 0: K = {0..7, 16..23}; lane-half 1: K = {8..15, 24..31}.
__device__ __forceinline__ v16h load_fragA(const _Float16* base, int ld) {
  const int lane = threadIdx.x & 31;
  const _Float16* p = base + (lane & 15) * ld + ((lane >> 4) << 3);
  h8 lo = *(const h8*)p;
  h8 hi = *(const h8*)(p + 16);
  v16h f;
#pragma unroll
  for (int i = 0; i < 8; ++i) { f[i] = lo[i]; f[i + 8] = hi[i]; }
  return f;
}

// B-fragment (32x16, K x N) built from row-major W[N][K] (B = W^T):
// lane n (n = lane&15) holds column n = W row n.
// Lane-half 0: K = {0..7, 16..23}; lane-half 1: K = {8..15, 24..31}.
__device__ __forceinline__ v16h load_fragB(const _Float16* base, int ld) {
  const int lane = threadIdx.x & 31;
  const _Float16* p = base + (lane & 15) * ld + ((lane >> 4) << 3);
  h8 lo = *(const h8*)p;
  h8 hi = *(const h8*)(p + 16);
  v16h f;
#pragma unroll
  for (int i = 0; i < 8; ++i) { f[i] = lo[i]; f[i + 8] = hi[i]; }
  return f;
}

// ---------------------------------------------------------------------------
// fp32 -> f16 convert (grid-stride)
// ---------------------------------------------------------------------------
__global__ void cvt_f32_to_f16(const float* __restrict__ in,
                               _Float16* __restrict__ out, int n) {
  int stride = gridDim.x * blockDim.x;
  for (int i = blockIdx.x * blockDim.x + threadIdx.x; i < n; i += stride)
    out[i] = (_Float16)in[i];
}

// ---------------------------------------------------------------------------
// TDM 2D tile load: tile_d1 rows x tile_d0 halfs from a row-major
// [dim1 x dim0-strided] f16 tensor. TDM pad feature inserts pad DWORDs into
// the LDS image so the LDS row stride matches the fragment loaders.
//   pad_interval code: pad fires every (1<<code) qwords of payload
//   pad_amount  code: (code+1) DWORDs of padding
// ---------------------------------------------------------------------------
#if USE_TDM
__device__ __forceinline__ void tdm_load_2d(const _Float16* gptr,
                                            unsigned int lds_off,
                                            int dim0, int dim1, int stride0,
                                            unsigned int tile_d0,
                                            unsigned int tile_d1,
                                            unsigned int pad_interval,
                                            unsigned int pad_amount) {
  unsigned long long ga = (unsigned long long)(uintptr_t)gptr;
  v4u g0;
  g0[0] = 1u;                                           // count=1, user mode
  g0[1] = lds_off;                                      // lds_addr (bytes)
  g0[2] = (unsigned int)ga;                             // global_addr[31:0]
  g0[3] = (unsigned int)((ga >> 32) & 0x1FFFFFFu) | (2u << 30);  // [56:32], type=2
  v8i g1;
  // data_size=1 (2B) | pad_enable | pad_interval | pad_amount
  g1[0] = (int)((1u << 16) | (1u << 20) | (pad_interval << 22) | (pad_amount << 25));
  g1[1] = (int)(((unsigned int)dim0 & 0xffffu) << 16);           // tensor_dim0 lo
  g1[2] = (int)(((unsigned int)dim0 >> 16) |
                (((unsigned int)dim1 & 0xffffu) << 16));         // dim0 hi | dim1 lo
  g1[3] = (int)(((unsigned int)dim1 >> 16) | (tile_d0 << 16));   // dim1 hi | tile_dim0
  g1[4] = (int)tile_d1;                                          // tile_dim1, tile_dim2=0
  g1[5] = (int)(unsigned int)stride0;                            // tensor_dim0_stride lo
  g1[6] = 0;
  g1[7] = 0;
  v4i z4 = {};
#if __clang_major__ >= 23
  v8i z8 = {};
  __builtin_amdgcn_tensor_load_to_lds(g0, g1, z4, z4, z8, 0);
#else
  __builtin_amdgcn_tensor_load_to_lds(g0, g1, z4, z4, 0);
#endif
}
#endif

// ---------------------------------------------------------------------------
// GEMM: C[M x N] (f32) = A[M x K] (f16 row-major) * B[N x K]^T (f16 row-major)
// Block tile 128x128, 8 waves (2x4), wave tile 64x32 (4x2 WMMA), BK=32.
// Double-buffered TDM staging: issue next tile's DMA, s_wait_tensorcnt 2 so
// the current tile is resident (TENSORcnt completes in order), then compute.
// ---------------------------------------------------------------------------
#define BM 128
#define BN 128
#define BK 32
#define LDS_STRIDE 40   // 32 + 8 pad halfs (matches TDM pad settings)

__global__ __launch_bounds__(256) void gemm_f16f32(
    const _Float16* __restrict__ A, const _Float16* __restrict__ B,
    float* __restrict__ C, int M, int N, int K) {
  __shared__ __align__(16) _Float16 As[2][BM * LDS_STRIDE];
  __shared__ __align__(16) _Float16 Bs[2][BN * LDS_STRIDE];

  const int tid  = threadIdx.x;
  const int lane = tid & 31;
  const int wave = tid >> 5;
  const int wm = wave >> 2;          // 0..1 : M offset wm*64
  const int wn = wave & 3;           // 0..3 : N offset wn*32
  const int m0 = blockIdx.y * BM;
  const int n0 = blockIdx.x * BN;

  v8f acc[4][2] = {};

  const int sr = tid >> 1;           // 0..127 staging row
  const int sc = (tid & 1) << 4;     // 0 or 16
  const int nsteps = K / BK;

#if USE_TDM
  if (wave == 0) {  // prologue: stage tile 0 into buffer 0
    tdm_load_2d(A + (size_t)m0 * K, (unsigned int)(uintptr_t)As[0], K, M, K, 32, BM, 3, 3);
    tdm_load_2d(B + (size_t)n0 * K, (unsigned int)(uintptr_t)Bs[0], K, N, K, 32, BN, 3, 3);
  }
#endif

  for (int i = 0; i < nsteps; ++i) {
    const int cur = i & 1;
    const int k0  = i * BK;
#if USE_TDM
    if (wave == 0) {
      if (i + 1 < nsteps) {  // overlap next tile's DMA with this tile's math
        tdm_load_2d(A + (size_t)m0 * K + k0 + BK,
                    (unsigned int)(uintptr_t)As[cur ^ 1], K, M, K, 32, BM, 3, 3);
        tdm_load_2d(B + (size_t)n0 * K + k0 + BK,
                    (unsigned int)(uintptr_t)Bs[cur ^ 1], K, N, K, 32, BN, 3, 3);
        __builtin_amdgcn_s_wait_tensorcnt(2);  // current tile resident
      } else {
        __builtin_amdgcn_s_wait_tensorcnt(0);
      }
    }
#else
    if (k0 + BK < K) {  // warm GL2 for the next tiles (global_prefetch_b8)
      __builtin_prefetch(A + (size_t)(m0 + sr) * K + k0 + BK + sc, 0, 1);
      __builtin_prefetch(B + (size_t)(n0 + sr) * K + k0 + BK + sc, 0, 1);
    }
    *(h8*)(As[cur] + sr * LDS_STRIDE + sc)     = *(const h8*)(A + (size_t)(m0 + sr) * K + k0 + sc);
    *(h8*)(As[cur] + sr * LDS_STRIDE + sc + 8) = *(const h8*)(A + (size_t)(m0 + sr) * K + k0 + sc + 8);
    *(h8*)(Bs[cur] + sr * LDS_STRIDE + sc)     = *(const h8*)(B + (size_t)(n0 + sr) * K + k0 + sc);
    *(h8*)(Bs[cur] + sr * LDS_STRIDE + sc + 8) = *(const h8*)(B + (size_t)(n0 + sr) * K + k0 + sc + 8);
#endif
    __syncthreads();

    v16h b0 = load_fragB(Bs[cur] + (wn * 32 + 0)  * LDS_STRIDE, LDS_STRIDE);
    v16h b1 = load_fragB(Bs[cur] + (wn * 32 + 16) * LDS_STRIDE, LDS_STRIDE);
#pragma unroll
    for (int mt = 0; mt < 4; ++mt) {
      v16h a = load_fragA(As[cur] + (wm * 64 + mt * 16) * LDS_STRIDE, LDS_STRIDE);
      acc[mt][0] = wmma32(a, b0, acc[mt][0]);
      acc[mt][1] = wmma32(a, b1, acc[mt][1]);
    }
    __syncthreads();  // last reads of buffer `cur` are now fenced for next DMA
  }

  const int hi8   = (lane >> 4) << 3;  // D layout: M = r + 8*(lane>=16)
  const int nlane = lane & 15;
#pragma unroll
  for (int mt = 0; mt < 4; ++mt)
#pragma unroll
    for (int nt = 0; nt < 2; ++nt) {
      int mbase = m0 + wm * 64 + mt * 16 + hi8;
      int nidx  = n0 + wn * 32 + nt * 16 + nlane;
#pragma unroll
      for (int r = 0; r < 8; ++r)
        C[(size_t)(mbase + r) * N + nidx] = acc[mt][nt][r];
    }
}

// ---------------------------------------------------------------------------
// RoPE (interleaved) + f32 -> f16
// ---------------------------------------------------------------------------
__global__ void rope_to_f16(const float* __restrict__ in,
                            const float* __restrict__ fc,
                            const float* __restrict__ fs,
                            _Float16* __restrict__ out, int nheads) {
  const int total  = T_SEQ * nheads * (HD / 2);
  const int stride = gridDim.x * blockDim.x;
  for (int i = blockIdx.x * blockDim.x + threadIdx.x; i < total; i += stride) {
    int pair = i % (HD / 2);
    int tmp  = i / (HD / 2);
    int h    = tmp % nheads;
    int t    = tmp / nheads;
    int d0   = pair * 2;
    size_t base = ((size_t)t * nheads + h) * HD + d0;
    float a = in[base], b = in[base + 1];
    out[base]     = (_Float16)(a * fc[t * HD + d0]     - b * fs[t * HD + d0]);
    out[base + 1] = (_Float16)(b * fc[t * HD + d0 + 1] + a * fs[t * HD + d0 + 1]);
  }
}

// ---------------------------------------------------------------------------
// Flash attention (causal, GQA rep=4), WMMA throughout.
// grid = (T/64, HQ), block = 128 (4 waves). Wave w owns q rows [qblk*64+w*16, +16).
// K tile staged by TDM (pad reproduces the 136-half LDK stride); V tile
// transposed manually by all threads in parallel with the DMA.
// ---------------------------------------------------------------------------
#define QB  64
#define KB  32
#define LDK 136
#define LDV 40

__global__ __launch_bounds__(128) void attn_kernel(
    const _Float16* __restrict__ q16,   // [T][HQ*HD]
    const _Float16* __restrict__ k16,   // [T][HKV*HD]
    const _Float16* __restrict__ v16,   // [T][HKV*HD]
    _Float16* __restrict__ y16) {       // [T][HQ*HD]
  __shared__ __align__(16) _Float16 Ks[KB * LDK];     // K tile, row-major [key][hd]
  __shared__ __align__(16) _Float16 Vts[HD * LDV];    // V tile, transposed [hd][key]
  __shared__ __align__(16) _Float16 Psh[4 * 16 * KB]; // per-wave P repack buffer

  const int tid   = threadIdx.x;
  const int lane  = tid & 31;
  const int wave  = tid >> 5;
  const int qblk  = blockIdx.x;
  const int h     = blockIdx.y;
  const int kh    = h >> 2;                 // GQA: 4 q-heads per kv-head
  const int qbase = qblk * QB + wave * 16;
  const int hi8   = (lane >> 4) << 3;
  const int nlane = lane & 15;
  const float k2  = 0.08838834764831845f * 1.4426950408889634f; // scale * log2(e)

  // Q fragments in registers: 4 chunks of K=32 over HD=128 (A-fragment layout).
  v16h qfrag[4];
  {
    const _Float16* p = q16 + (size_t)(qbase + nlane) * QKD + h * HD;
#pragma unroll
    for (int c = 0; c < 4; ++c) {
      const _Float16* pc = p + c * 32 + hi8;
      h8 lo = *(const h8*)pc;
      h8 hi = *(const h8*)(pc + 16);
      v16h f;
#pragma unroll
      for (int i = 0; i < 8; ++i) { f[i] = lo[i]; f[i + 8] = hi[i]; }
      qfrag[c] = f;
    }
  }

  v8f O[8] = {};
  float mrow[8], lrow[8];
#pragma unroll
  for (int r = 0; r < 8; ++r) { mrow[r] = -1e30f; lrow[r] = 0.0f; }

  const int nkt = (qblk * QB + QB) / KB;  // key tiles needed (causal)
  for (int kt = 0; kt < nkt; ++kt) {
    const int ktbase = kt * KB;

    // Stage K tile (TDM if available) and V tile transposed (manual).
#if USE_TDM
    if (wave == 0) {
      // 32 rows x 128 halfs, row stride KVD; pad 4 DW every 32 qwords -> LDK=136
      tdm_load_2d(k16 + (size_t)ktbase * KVD + kh * HD,
                  (unsigned int)(uintptr_t)Ks, KVD, T_SEQ, KVD, 128, KB, 5, 3);
    }
#endif
    {
      const int krow = tid & 31;
      const int cg   = (tid >> 5) * 32;
#if !USE_TDM
      const _Float16* ksrc = k16 + (size_t)(ktbase + krow) * KVD + kh * HD + cg;
      _Float16* kdst = Ks + krow * LDK + cg;
#pragma unroll
      for (int u = 0; u < 4; ++u)
        *(h8*)(kdst + u * 8) = *(const h8*)(ksrc + u * 8);
#endif
      const _Float16* vsrc = v16 + (size_t)(ktbase + krow) * KVD + kh * HD + cg;
#pragma unroll
      for (int u = 0; u < 32; ++u)
        Vts[(cg + u) * LDV + krow] = vsrc[u];
    }
#if USE_TDM
    if (wave == 0) __builtin_amdgcn_s_wait_tensorcnt(0);
#endif
    __syncthreads();

    if (ktbase <= qbase + 15) {  // wave-uniform skip of fully masked tiles
      // S = Q * K^T : two 16x16 D tiles over the 32 keys
      v8f s[2] = {};
#pragma unroll
      for (int c = 0; c < 4; ++c) {
        v16h b0 = load_fragB(Ks + 0 * LDK + c * 32, LDK);
        v16h b1 = load_fragB(Ks + 16 * LDK + c * 32, LDK);
        s[0] = wmma32(qfrag[c], b0, s[0]);
        s[1] = wmma32(qfrag[c], b1, s[1]);
      }

      float p0[8], p1[8], alpha[8];
#pragma unroll
      for (int r = 0; r < 8; ++r) {
        const int qrow = qbase + hi8 + r;
        float a0 = s[0][r] * k2;
        float a1 = s[1][r] * k2;
        if (ktbase + nlane > qrow)      a0 = -1e30f;
        if (ktbase + 16 + nlane > qrow) a1 = -1e30f;
        float tm = fmaxf(a0, a1);
        tm = fmaxf(tm, __shfl_xor(tm, 1, 32));
        tm = fmaxf(tm, __shfl_xor(tm, 2, 32));
        tm = fmaxf(tm, __shfl_xor(tm, 4, 32));
        tm = fmaxf(tm, __shfl_xor(tm, 8, 32));
        float mnew = fmaxf(mrow[r], tm);
        alpha[r] = exp2f(mrow[r] - mnew);
        float e0 = exp2f(a0 - mnew);
        float e1 = exp2f(a1 - mnew);
        float rs = e0 + e1;
        rs += __shfl_xor(rs, 1, 32);
        rs += __shfl_xor(rs, 2, 32);
        rs += __shfl_xor(rs, 4, 32);
        rs += __shfl_xor(rs, 8, 32);
        lrow[r] = lrow[r] * alpha[r] + rs;
        mrow[r] = mnew;
        p0[r] = e0;
        p1[r] = e1;
      }

      // Repack P (D layout) -> A layout through per-wave LDS buffer.
      _Float16* pbase = Psh + wave * 16 * KB;
#pragma unroll
      for (int r = 0; r < 8; ++r) {
        pbase[(hi8 + r) * KB + nlane]      = (_Float16)p0[r];
        pbase[(hi8 + r) * KB + 16 + nlane] = (_Float16)p1[r];
      }
      v16h pf = load_fragA(pbase, KB);

      // O = O*alpha + P * V  (V^T rows in Vts serve as the B operand)
#pragma unroll
      for (int nc = 0; nc < 8; ++nc) {
        v16h vf = load_fragB(Vts + nc * 16 * LDV, LDV);
#pragma unroll
        for (int r = 0; r < 8; ++r) O[nc][r] *= alpha[r];
        O[nc] = wmma32(pf, vf, O[nc]);
      }
    }
    __syncthreads();
  }

  // Normalize and store y16.
  float linv[8];
#pragma unroll
  for (int r = 0; r < 8; ++r) linv[r] = 1.0f / lrow[r];
#pragma unroll
  for (int nc = 0; nc < 8; ++nc) {
#pragma unroll
    for (int r = 0; r < 8; ++r) {
      const int qrow = qbase + hi8 + r;
      y16[(size_t)qrow * QKD + h * HD + nc * 16 + nlane] =
          (_Float16)(O[nc][r] * linv[r]);
    }
  }
}

// ---------------------------------------------------------------------------
// Host-side orchestration
// ---------------------------------------------------------------------------
extern "C" void kernel_launch(void* const* d_in, const int* in_sizes, int n_in,
                              void* d_out, int out_size, void* d_ws, size_t ws_size,
                              hipStream_t stream) {
  (void)in_sizes; (void)n_in; (void)out_size; (void)ws_size;

  const float* x  = (const float*)d_in[0];
  const float* wq = (const float*)d_in[1];
  const float* wk = (const float*)d_in[2];
  const float* wv = (const float*)d_in[3];
  const float* wo = (const float*)d_in[4];
  const float* fc = (const float*)d_in[5];
  const float* fs = (const float*)d_in[6];
  float* out = (float*)d_out;

  const int NX  = T_SEQ * DIM;        // 6291456
  const int NWQ = QKD * DIM;          // 12582912
  const int NWK = KVD * DIM;          // 3145728
  const int NWO = DIM * QKD;          // 12582912
  const int NQ  = T_SEQ * QKD;        // 8388608
  const int NK  = T_SEQ * KVD;        // 2097152

  char* ws = (char*)d_ws;
  size_t off = 0;
  auto take = [&](size_t bytes) -> char* {
    char* p = ws + off;
    off += (bytes + 255) & ~(size_t)255;
    return p;
  };
  _Float16* x16  = (_Float16*)take((size_t)NX  * 2);
  _Float16* wq16 = (_Float16*)take((size_t)NWQ * 2);
  _Float16* wk16 = (_Float16*)take((size_t)NWK * 2);
  _Float16* wv16 = (_Float16*)take((size_t)NWK * 2);
  _Float16* wo16 = (_Float16*)take((size_t)NWO * 2);
  float*    qf   = (float*)   take((size_t)NQ  * 4);
  float*    kf   = (float*)   take((size_t)NK  * 4);
  float*    vf   = (float*)   take((size_t)NK  * 4);
  _Float16* q16  = (_Float16*)take((size_t)NQ  * 2);
  _Float16* k16  = (_Float16*)take((size_t)NK  * 2);
  _Float16* v16  = (_Float16*)take((size_t)NK  * 2);
  _Float16* y16  = (_Float16*)take((size_t)NQ  * 2);

  // 1) fp32 -> f16 conversions
  cvt_f32_to_f16<<<2048, 256, 0, stream>>>(x,  x16,  NX);
  cvt_f32_to_f16<<<2048, 256, 0, stream>>>(wq, wq16, NWQ);
  cvt_f32_to_f16<<<2048, 256, 0, stream>>>(wk, wk16, NWK);
  cvt_f32_to_f16<<<2048, 256, 0, stream>>>(wv, wv16, NWK);
  cvt_f32_to_f16<<<2048, 256, 0, stream>>>(wo, wo16, NWO);

  // 2) QKV projections (WMMA GEMMs): C = A * W^T
  gemm_f16f32<<<dim3(QKD / BN, T_SEQ / BM), 256, 0, stream>>>(x16, wq16, qf, T_SEQ, QKD, DIM);
  gemm_f16f32<<<dim3(KVD / BN, T_SEQ / BM), 256, 0, stream>>>(x16, wk16, kf, T_SEQ, KVD, DIM);
  gemm_f16f32<<<dim3(KVD / BN, T_SEQ / BM), 256, 0, stream>>>(x16, wv16, vf, T_SEQ, KVD, DIM);

  // 3) RoPE + convert to f16 (v is plain convert)
  rope_to_f16<<<2048, 256, 0, stream>>>(qf, fc, fs, q16, HQ);
  rope_to_f16<<<2048, 256, 0, stream>>>(kf, fc, fs, k16, HKV);
  cvt_f32_to_f16<<<2048, 256, 0, stream>>>(vf, v16, NK);

  // 4) causal flash attention (input_pos == 0; cache is zero & masked beyond T)
  attn_kernel<<<dim3(T_SEQ / QB, HQ), 128, 0, stream>>>(q16, k16, v16, y16);

  // 5) output projection -> fp32 d_out
  gemm_f16f32<<<dim3(DIM / BN, T_SEQ / BM), 256, 0, stream>>>(y16, wo16, out, T_SEQ, DIM, QKD);
}
